// AdaptiveConv_824633721409
// MI455X (gfx1250) — matrix operand order repaired and checked by reading, not compiled, not run on hardware
//
#include <hip/hip_runtime.h>
#include <hip/hip_bf16.h>

typedef __attribute__((ext_vector_type(16))) _Float16 v16h;
typedef __attribute__((ext_vector_type(8)))  _Float16 v8h;
typedef __attribute__((ext_vector_type(8)))  float    v8f;

#define NBATCH 16
#define CIN    256
#define COUT   256
#define HH     32
#define WW     32
#define HW     1024          // HH*WW
#define KTAPS  9
#define KDIM   2304          // KTAPS*CIN
#define MTOT   16384         // NBATCH*HW
#define KSTEPS (KDIM / 32)   // 72 (even)
#define BPAD   40            // padded LDS row stride (halves): 20 dwords -> conflict-free

// ---------------------------------------------------------------------------
// Kernel 0: repack weight[o][c][k] (f32) -> Wr[o][k*256+c] (f16), row-major.
// ---------------------------------------------------------------------------
__global__ void __launch_bounds__(256)
pack_weight_k(const float* __restrict__ w, _Float16* __restrict__ Wr) {
    int t = blockIdx.x * 256 + threadIdx.x;      // t < COUT*KDIM (exact grid)
    int o  = t / KDIM;
    int kc = t - o * KDIM;
    int k  = kc >> 8;
    int c  = kc & 255;
    Wr[(size_t)o * KDIM + kc] = (_Float16)w[((size_t)o * CIN + c) * KTAPS + k];
}

// ---------------------------------------------------------------------------
// Kernel 1: bilinear sampling -> f16 im2col A[m][k*256+c], m = n*HW + p.
// One block per (n,p); lane = channel -> 64B coalesced f16 stores.
// x (16MB) is L2-resident (192MB L2), so the strided corner gathers hit L2.
// ---------------------------------------------------------------------------
__global__ void __launch_bounds__(256)
sample_im2col_k(const float* __restrict__ x, const float* __restrict__ off,
                _Float16* __restrict__ A) {
    const int bp = blockIdx.x;               // n*HW + p  (exact grid = MTOT)
    const int n  = bp >> 10;
    const int p  = bp & (HW - 1);
    const int yy = p >> 5;
    const int xx = p & 31;
    const int c  = threadIdx.x;

    const float* xplane = x + ((size_t)(n * CIN + c)) * HW;   // channel plane
    const float* offp   = off + (size_t)bp * (2 * KTAPS);
    _Float16*    arow   = A + (size_t)bp * KDIM;

    #pragma unroll
    for (int k = 0; k < KTAPS; ++k) {
        float dy = offp[2 * k];                     // broadcast across lanes
        float dx = offp[2 * k + 1];
        float py = (float)(yy + (k / 3) - 1) + dy;
        float px = (float)(xx + (k % 3) - 1) + dx;
        float y0f = floorf(py), x0f = floorf(px);
        float ly = py - y0f,    lx = px - x0f;
        int y0 = (int)y0f, x0 = (int)x0f;

        float acc = 0.f;
        #pragma unroll
        for (int cy = 0; cy < 2; ++cy) {
            #pragma unroll
            for (int cx = 0; cx < 2; ++cx) {
                int yi = y0 + cy, xi = x0 + cx;
                float wgt = (cy ? ly : 1.f - ly) * (cx ? lx : 1.f - lx);
                bool ok = (yi >= 0) && (yi < HH) && (xi >= 0) && (xi < WW);
                int yc = min(max(yi, 0), HH - 1);
                int xc = min(max(xi, 0), WW - 1);
                float xv = xplane[yc * WW + xc];
                acc += ok ? wgt * xv : 0.f;
            }
        }
        arow[(k << 8) + c] = (_Float16)acc;
    }
}

// ---------------------------------------------------------------------------
// Kernel 2: GEMM  out[m][o] = A[m][:] . Wr[o][:]  via v_wmma_f32_16x16x32_f16.
// Block = 8 waves = 128(M) x 64(N); B tile (32K x 64N) staged in LDS,
// double-buffered. K-loop: branch-free x2-unrolled steady state over the
// first 70 K-steps + peeled 2-step epilogue -> no accumulator copies.
// ---------------------------------------------------------------------------
__global__ void __launch_bounds__(256)
gemm_wmma_k(const _Float16* __restrict__ A, const _Float16* __restrict__ Wr,
            float* __restrict__ out) {
    __shared__ _Float16 lb[2][64 * BPAD];     // 10 KB double-buffered B stage

    const int tid  = threadIdx.x;
    const int wavw = tid >> 5;                // 0..7 : wave's 16-row sub-block
    const int lane = tid & 31;
    const int lh   = lane & 15;
    const bool hi  = lane >= 16;

    const int mblk  = blockIdx.x >> 2;        // 0..127 : 128-row block
    const int nb4   = blockIdx.x & 3;         // 0..3   : 64-col group
    const int m0    = (mblk << 7) + (wavw << 4);
    const int obase = nb4 << 6;               // nb4*64

    // B staging: thread t covers row so = t>>2, 8 halves at sj = (t&3)*8
    const int so = tid >> 2;
    const int sj = (tid & 3) << 3;
    const _Float16* bsrc = Wr + (size_t)(obase + so) * KDIM + sj;
    const int sdst = so * BPAD + sj;          // halves offset within a buffer

    // A fragment pointer (pre-biased per documented interleaved layout:
    // lanes 0-15 hold K{0..7,16..23}, lanes 16-31 hold K{8..15,24..31})
    const _Float16* arow = A + (size_t)(m0 + lh) * KDIM + (hi ? 8 : 0);

    // B fragment base offset in LDS (halves); lanes 0-15: K{0..7,16..23}, 16-31: K{8..15,24..31}
    const int boff = lh * BPAD + (hi ? 8 : 0);

    union U { v16h v; v8h h[2]; };

    v8f acc0 = {}, acc1 = {}, acc2 = {}, acc3 = {};

    // WMMA half-step against lb[buf]; caller already issued next-tile loads.
    auto mm_step = [&](const U& afrag, int buf) {
        const _Float16* lbase = &lb[buf][0];
        U b0, b1, b2, b3;
        b0.h[0] = *(const v8h*)(lbase + boff);
        b0.h[1] = *(const v8h*)(lbase + boff + 16);
        b1.h[0] = *(const v8h*)(lbase + 16 * BPAD + boff);
        b1.h[1] = *(const v8h*)(lbase + 16 * BPAD + boff + 16);
        b2.h[0] = *(const v8h*)(lbase + 32 * BPAD + boff);
        b2.h[1] = *(const v8h*)(lbase + 32 * BPAD + boff + 16);
        b3.h[0] = *(const v8h*)(lbase + 48 * BPAD + boff);
        b3.h[1] = *(const v8h*)(lbase + 48 * BPAD + boff + 16);
        acc0 = __builtin_amdgcn_wmma_f32_16x16x32_f16(false, afrag.v, false, b0.v,
                                                      (short)0, acc0, false, false);
        acc1 = __builtin_amdgcn_wmma_f32_16x16x32_f16(false, afrag.v, false, b1.v,
                                                      (short)0, acc1, false, false);
        acc2 = __builtin_amdgcn_wmma_f32_16x16x32_f16(false, afrag.v, false, b2.v,
                                                      (short)0, acc2, false, false);
        acc3 = __builtin_amdgcn_wmma_f32_16x16x32_f16(false, afrag.v, false, b3.v,
                                                      (short)0, acc3, false, false);
    };

    // ---- prologue: stage K-step 0 into buffer 0, preload A fragment 0
    v8h greg = *(const v8h*)(bsrc);
    U a0, a1;
    a0.h[0] = *(const v8h*)(arow);
    a0.h[1] = *(const v8h*)(arow + 16);
    *(v8h*)(&lb[0][sdst]) = greg;
    __syncthreads();

    // ---- branch-free steady state: double-steps over K-steps 0..69.
    // After each iteration, lb[0]/a0 hold step i+2.
    for (int i = 0; i < KSTEPS - 2; i += 2) {
        const int kb = i << 5;

        // even half: consume a0 / lb[0]; prefetch step i+1
        a1.h[0] = *(const v8h*)(arow + kb + 32);
        a1.h[1] = *(const v8h*)(arow + kb + 48);
        greg    = *(const v8h*)(bsrc + kb + 32);
        __builtin_prefetch(arow + kb + 256, 0, 1);       // global_prefetch_b8

        mm_step(a0, 0);

        *(v8h*)(&lb[1][sdst]) = greg;
        __syncthreads();

        // odd half: consume a1 / lb[1]; prefetch step i+2 (<= step 70, in range)
        a0.h[0] = *(const v8h*)(arow + kb + 64);
        a0.h[1] = *(const v8h*)(arow + kb + 80);
        greg    = *(const v8h*)(bsrc + kb + 64);

        mm_step(a1, 1);

        *(v8h*)(&lb[0][sdst]) = greg;
        __syncthreads();
    }

    // ---- peeled epilogue: K-steps 70 and 71
    {
        const int kb = (KSTEPS - 2) << 5;                // step 70 base
        a1.h[0] = *(const v8h*)(arow + kb + 32);         // A frag of step 71
        a1.h[1] = *(const v8h*)(arow + kb + 48);
        greg    = *(const v8h*)(bsrc + kb + 32);         // B tile of step 71

        mm_step(a0, 0);                                  // step 70

        *(v8h*)(&lb[1][sdst]) = greg;
        __syncthreads();

        mm_step(a1, 1);                                  // step 71 (no staging)
    }

    // C/D layout: VGPR r -> row M = r + (hi?8:0); lane%16 -> column N.
    #pragma unroll
    for (int r = 0; r < 8; ++r) {
        int mrow = m0 + r + (hi ? 8 : 0);
        int nimg = mrow >> 10;
        int p    = mrow & (HW - 1);
        size_t base = ((size_t)nimg * COUT) * HW + p;
        out[base + (size_t)(obase +  0 + lh) * HW] = acc0[r];
        out[base + (size_t)(obase + 16 + lh) * HW] = acc1[r];
        out[base + (size_t)(obase + 32 + lh) * HW] = acc2[r];
        out[base + (size_t)(obase + 48 + lh) * HW] = acc3[r];
    }
}

// ---------------------------------------------------------------------------
extern "C" void kernel_launch(void* const* d_in, const int* in_sizes, int n_in,
                              void* d_out, int out_size, void* d_ws, size_t ws_size,
                              hipStream_t stream) {
    const float* x   = (const float*)d_in[0];   // (16,256,32,32)
    const float* off = (const float*)d_in[1];   // (16,1024,18)
    const float* w   = (const float*)d_in[2];   // (256,256,3,3)
    float* out = (float*)d_out;                 // (16,256,32,32)

    _Float16* Abuf = (_Float16*)d_ws;                                    // 75.5 MB
    _Float16* Wr   = (_Float16*)((char*)d_ws + (size_t)MTOT * KDIM * 2); // 1.2 MB

    // weight repack: COUT*KDIM / 256 = 2304 blocks (exact)
    pack_weight_k<<<(COUT * KDIM) / 256, 256, 0, stream>>>(w, Wr);

    // im2col sampling: one block per (n,p)
    sample_im2col_k<<<MTOT, 256, 0, stream>>>(x, off, Abuf);

    // GEMM: 128 M-blocks (128 rows) x 4 N-groups = 512 blocks of 8 waves
    gemm_wmma_k<<<512, 256, 0, stream>>>(Abuf, Wr, out);
}
